// GatNet_15358803050747
// MI455X (gfx1250) — matrix operand order repaired and checked by reading, not compiled, not run on hardware
//
#include <hip/hip_runtime.h>
#include <math.h>

// GAT on gfx1250: WMMA f16->f32 for node transforms, atomic scatter for
// edge softmax/aggregation (L2-resident destinations on MI455X's 192MB L2).

typedef __attribute__((ext_vector_type(16))) _Float16 v16h;
typedef __attribute__((ext_vector_type(8)))  float    v8f;

#define HC 64   // hidden channels = H*C
#define NH 8    // heads
#define NC 8    // channels per head

// ---------- ordered-uint encoding for float atomicMax ----------
__device__ __forceinline__ unsigned f2ord(float f) {
  unsigned u = __float_as_uint(f);
  return (u & 0x80000000u) ? ~u : (u | 0x80000000u);
}
__device__ __forceinline__ float ord2f(unsigned u) {
  return (u & 0x80000000u) ? __uint_as_float(u & 0x7FFFFFFFu)
                           : __uint_as_float(~u);
}
#define ORD_NEG_INF 0x007FFFFFu   // f2ord(-inf)

// ---------- layer 1 transform: F_in = 1 ----------
__global__ void l1_transform(const float* __restrict__ x,
                             const float* __restrict__ W1,
                             float* __restrict__ hout, int N) {
  int i = blockIdx.x * blockDim.x + threadIdx.x;
  if (i >= N * HC) return;
  int n = i >> 6, f = i & 63;
  hout[i] = x[n] * W1[f];
}

// ---------- WMMA node transform: H = act[N,64] @ W[64,64] ----------
// One wave computes a 16-row x 64-col tile: 4 N-tiles x 2 K-chunks of
// v_wmma_f32_16x16x32_f16.  f32 inputs converted to f16 in-register.
__global__ __launch_bounds__(256) void gemm_nodes_wmma(
    const float* __restrict__ act, const float* __restrict__ W,
    float* __restrict__ hout, int nTiles, int N) {
  int wave = blockIdx.x * (blockDim.x >> 5) + (threadIdx.x >> 5);
  if (wave >= nTiles) return;                 // wave-uniform: EXEC stays full
  int lane = threadIdx.x & 31;
  int half = lane >> 4;                       // 0: lanes 0-15, 1: lanes 16-31
  int mloc = lane & 15;

  int tileBase = wave * 16;
  bool fullTile = (tileBase + 16 <= N);       // wave-uniform fast path

  int rowA = tileBase + mloc;
  if (rowA >= N) rowA = N - 1;                // clamp for tail-tile loads
  const float* arow = act + (size_t)rowA * HC;

  // Pack A (16x32 f16 per k-chunk) per ISA layout:
  //   elem j<8  -> K = kb + 8*half + j
  //   elem j>=8 -> K = kb + 16 + 8*half + (j-8)
  v16h aK[2];
#pragma unroll
  for (int kc = 0; kc < 2; ++kc) {
    int kb = kc * 32;
#pragma unroll
    for (int j = 0; j < 8; ++j)
      aK[kc][j] = (_Float16)arow[kb + 8 * half + j];
#pragma unroll
    for (int j = 0; j < 8; ++j)
      aK[kc][8 + j] = (_Float16)arow[kb + 16 + 8 * half + j];
  }

#pragma unroll
  for (int nt = 0; nt < 4; ++nt) {
    int ncol = nt * 16 + mloc;
    v8f c = {};
#pragma unroll
    for (int kc = 0; kc < 2; ++kc) {
      int kb = kc * 32;
      // Pack B (32x16 f16): elem j<8 -> K = kb + 8*half + j, j>=8 -> K = kb + 16 + 8*half + (j-8), col = ncol
      v16h b;
#pragma unroll
      for (int j = 0; j < 16; ++j)
        b[j] = (_Float16)W[(size_t)(kb + ((j < 8) ? (8 * half + j) : (16 + 8 * half + (j - 8)))) * HC + ncol];
      c = __builtin_amdgcn_wmma_f32_16x16x32_f16(
          false, aK[kc], false, b, (short)0, c, false, false);
    }
    // D layout: elem r -> row = tileBase + r + 8*half, col = ncol
    float* dcol = hout + ((size_t)(tileBase + 8 * half)) * HC + ncol;
    if (fullTile) {
      // uniform fast path: straight-line stores, single base address
#pragma unroll
      for (int r = 0; r < 8; ++r)
        dcol[(size_t)r * HC] = c[r];
    } else {
#pragma unroll
      for (int r = 0; r < 8; ++r) {
        int row = tileBase + 8 * half + r;
        if (row < N) dcol[(size_t)r * HC] = c[r];
      }
    }
  }
}

// ---------- per-node attention scores ----------
__global__ void scores_kernel(const float* __restrict__ h,
                              const float* __restrict__ a_src,
                              const float* __restrict__ a_dst,
                              float* __restrict__ s, float* __restrict__ d,
                              int N) {
  int i = blockIdx.x * blockDim.x + threadIdx.x;
  if (i >= N * NH) return;
  int n = i >> 3, head = i & 7;
  const float* hr = h + (size_t)n * HC + head * NC;
  float ss = 0.f, dd = 0.f;
#pragma unroll
  for (int c = 0; c < NC; ++c) {
    ss += hr[c] * a_src[head * NC + c];
    dd += hr[c] * a_dst[head * NC + c];
  }
  s[i] = ss;
  d[i] = dd;
}

// ---------- init emax / esum / agg ----------
__global__ void init_kernel(unsigned* __restrict__ emaxU,
                            float* __restrict__ esum,
                            float* __restrict__ agg, int N) {
  int i = blockIdx.x * blockDim.x + threadIdx.x;
  if (i < N * HC) agg[i] = 0.f;
  if (i < N * NH) { emaxU[i] = ORD_NEG_INF; esum[i] = 0.f; }
}

// ---------- edge phase 1: leaky-relu score + segment max ----------
__global__ void edge_phase1(const int* __restrict__ ei,
                            const float* __restrict__ s,
                            const float* __restrict__ d,
                            float* __restrict__ ebuf,
                            unsigned* __restrict__ emaxU, int E, int N) {
  int e = blockIdx.x * blockDim.x + threadIdx.x;
  if (e >= E + N) return;
  int src, dst;
  if (e < E) { src = ei[e]; dst = ei[E + e]; } else { src = dst = e - E; }
#pragma unroll
  for (int h = 0; h < NH; ++h) {
    float v = s[src * NH + h] + d[dst * NH + h];
    v = (v >= 0.f) ? v : 0.2f * v;            // leaky_relu 0.2
    ebuf[(size_t)e * NH + h] = v;
    atomicMax(&emaxU[dst * NH + h], f2ord(v));
  }
}

// ---------- edge phase 2: exp(e - max) + segment sum ----------
__global__ void edge_phase2(const int* __restrict__ ei,
                            const unsigned* __restrict__ emaxU,
                            float* __restrict__ ebuf,
                            float* __restrict__ esum, int E, int N) {
  int t = blockIdx.x * blockDim.x + threadIdx.x;
  if (t >= (E + N) * NH) return;
  int e = t >> 3, h = t & 7;
  int dst = (e < E) ? ei[E + e] : (e - E);
  float w = __expf(ebuf[t] - ord2f(emaxU[dst * NH + h]));
  ebuf[t] = w;
  atomicAdd(&esum[dst * NH + h], w);
}

// ---------- edge phase 3: alpha-weighted message scatter-add ----------
__global__ void edge_phase3(const int* __restrict__ ei,
                            const float* __restrict__ ebuf,
                            const float* __restrict__ esum,
                            const float* __restrict__ hfeat,
                            float* __restrict__ agg, int E, int N) {
  int t = blockIdx.x * blockDim.x + threadIdx.x;
  if (t >= (E + N) * NH) return;
  int e = t >> 3, h = t & 7;
  int src, dst;
  if (e < E) { src = ei[e]; dst = ei[E + e]; } else { src = dst = e - E; }
  float alpha = ebuf[t] / (esum[dst * NH + h] + 1e-16f);
  const float* hs = hfeat + (size_t)src * HC + h * NC;
  float* ad = agg + (size_t)dst * HC + h * NC;
#pragma unroll
  for (int c = 0; c < NC; ++c)
    atomicAdd(&ad[c], hs[c] * alpha);
}

// ---------- bias + ELU ----------
__global__ void bias_elu(const float* __restrict__ agg,
                         const float* __restrict__ b,
                         float* __restrict__ act, int N) {
  int i = blockIdx.x * blockDim.x + threadIdx.x;
  if (i >= N * HC) return;
  float v = agg[i] + b[i & 63];
  act[i] = (v > 0.f) ? v : (__expf(v) - 1.f);
}

// ---------- final FC 64 -> 1 ----------
__global__ void final_fc(const float* __restrict__ act,
                         const float* __restrict__ fc_w,
                         const float* __restrict__ fc_b,
                         float* __restrict__ out, int N) {
  int n = blockIdx.x * blockDim.x + threadIdx.x;
  if (n >= N) return;
  const float* r = act + (size_t)n * HC;
  float acc = 0.f;
#pragma unroll
  for (int f = 0; f < HC; ++f) acc += r[f] * fc_w[f];
  out[n] = acc + fc_b[0];
}

extern "C" void kernel_launch(void* const* d_in, const int* in_sizes, int n_in,
                              void* d_out, int out_size, void* d_ws, size_t ws_size,
                              hipStream_t stream) {
  const int N = in_sizes[0];
  const int E = in_sizes[1] / 2;
  const int Etot = E + N;

  const float* x  = (const float*)d_in[0];
  const int*   ei = (const int*)d_in[1];
  const float* W[3]   = {(const float*)d_in[2],  (const float*)d_in[6],  (const float*)d_in[10]};
  const float* asr[3] = {(const float*)d_in[3],  (const float*)d_in[7],  (const float*)d_in[11]};
  const float* ads[3] = {(const float*)d_in[4],  (const float*)d_in[8],  (const float*)d_in[12]};
  const float* bia[3] = {(const float*)d_in[5],  (const float*)d_in[9],  (const float*)d_in[13]};
  const float* fcw = (const float*)d_in[14];
  const float* fcb = (const float*)d_in[15];

  // workspace carve-out (256B aligned slices)
  char*  ws  = (char*)d_ws;
  size_t off = 0;
  auto carve = [&](size_t bytes) -> void* {
    off = (off + 255) & ~(size_t)255;
    void* p = ws + off;
    off += bytes;
    return p;
  };
  float*    act   = (float*)   carve((size_t)N * HC * 4);
  float*    hbuf  = (float*)   carve((size_t)N * HC * 4);
  float*    sS    = (float*)   carve((size_t)N * NH * 4);
  float*    sD    = (float*)   carve((size_t)N * NH * 4);
  unsigned* emaxU = (unsigned*)carve((size_t)N * NH * 4);
  float*    esum  = (float*)   carve((size_t)N * NH * 4);
  float*    agg   = (float*)   carve((size_t)N * HC * 4);
  float*    ebuf  = (float*)   carve((size_t)Etot * NH * 4);
  (void)ws_size;

  const int TB = 256;
  const int nTiles   = (N + 15) / 16;
  const int gGemm    = (nTiles + 7) / 8;          // 8 waves per block
  const int gNodeHC  = (N * HC + TB - 1) / TB;
  const int gNodeNH  = (N * NH + TB - 1) / TB;
  const int gEdge    = (Etot + TB - 1) / TB;
  const int gEdgeNH  = (Etot * NH + TB - 1) / TB;
  const int gNode    = (N + TB - 1) / TB;

  for (int l = 0; l < 3; ++l) {
    // 1) node transform
    if (l == 0)
      l1_transform<<<gNodeHC, TB, 0, stream>>>(x, W[0], hbuf, N);
    else
      gemm_nodes_wmma<<<gGemm, TB, 0, stream>>>(act, W[l], hbuf, nTiles, N);
    // 2) attention scores
    scores_kernel<<<gNodeNH, TB, 0, stream>>>(hbuf, asr[l], ads[l], sS, sD, N);
    // 3) reset segment buffers
    init_kernel<<<gNodeHC, TB, 0, stream>>>(emaxU, esum, agg, N);
    // 4) edge softmax + aggregation
    edge_phase1<<<gEdge,   TB, 0, stream>>>(ei, sS, sD, ebuf, emaxU, E, N);
    edge_phase2<<<gEdgeNH, TB, 0, stream>>>(ei, emaxU, ebuf, esum, E, N);
    edge_phase3<<<gEdgeNH, TB, 0, stream>>>(ei, ebuf, esum, hbuf, agg, E, N);
    // 5) bias + ELU -> next layer activation
    bias_elu<<<gNodeHC, TB, 0, stream>>>(agg, bia[l], act, N);
  }
  final_fc<<<gNode, TB, 0, stream>>>(act, fcw, fcb, (float*)d_out, N);
}
